// DynamicGraph_6373731467945
// MI455X (gfx1250) — matrix operand order repaired and checked by reading, not compiled, not run on hardware
//
#include <hip/hip_runtime.h>

// ---------------------------------------------------------------------------
// Problem constants (match reference)
// ---------------------------------------------------------------------------
#define BB     8
#define NN     2048
#define DD     1024
#define DKK    256
#define TOPK   32
#define MTOT   (BB * NN)          // 16384 rows for projections
#define INV_SCALE (1.0f / 16.0f)  // 1/sqrt(DK)

typedef __attribute__((ext_vector_type(16))) __bf16       v16bf;
typedef __attribute__((ext_vector_type(8)))  float        v8f;
typedef __attribute__((ext_vector_type(4)))  unsigned int v4u;
typedef __attribute__((ext_vector_type(8)))  unsigned int v8u;

union FragU { v16bf bf; v4u u[2]; };

__device__ __forceinline__ unsigned short f32_to_bf16(float f) {
  unsigned int u = __float_as_uint(f);
  u += 0x7FFFu + ((u >> 16) & 1u);     // round-to-nearest-even
  return (unsigned short)(u >> 16);
}

// monotone map: float bits -> unsigned key (descending float == descending key)
__device__ __forceinline__ unsigned int order_key(float v) {
  unsigned int u = __float_as_uint(v);
  return (u & 0x80000000u) ? ~u : (u | 0x80000000u);
}

// A-fragment (16x32 bf16, row-major source): lane<16: M=lane, K in {0..7,16..23};
// lane>=16: M=lane-16, K in {8..15,24..31}.  (ISA 7.12.2)
__device__ __forceinline__ v16bf load_a_frag(const unsigned short* base, int stride) {
  const int lane = threadIdx.x & 31;
  const int r = lane & 15;
  const int h = lane >> 4;
  const unsigned short* p = base + r * stride + h * 8;
  FragU f;
  f.u[0] = *(const v4u*)(p);        // K = h*8 .. h*8+7
  f.u[1] = *(const v4u*)(p + 16);   // K = 16 + h*8 .. 16 + h*8+7
  return f.bf;
}

// B-fragment (32x16 bf16), source row-major over N with K contiguous (i.e. the
// N x K "transposed" operand): lane<16: N=lane, K in {0..7,16..23}; lane>=16: N=lane-16, K in {8..15,24..31}.
__device__ __forceinline__ v16bf load_b_frag(const unsigned short* base, int stride) {
  const int lane = threadIdx.x & 31;
  const int r = lane & 15;
  const int h = lane >> 4;
  const unsigned short* p = base + r * stride + h * 8;
  FragU f;
  f.u[0] = *(const v4u*)(p);
  f.u[1] = *(const v4u*)(p + 16);
  return f.bf;
}

// ---------------------------------------------------------------------------
// TDM: issue a 2-D bf16 tile load Global->LDS (one instruction per block,
// wave-scalar; tracked by TENSORcnt).  CDNA5 ISA/§8.
//   rows      : tile_dim1 (# rows)
//   rowlen    : tile_dim0 (elements per row, == tensor_dim0 == dim0 stride)
//   pad_iv    : pad_interval code (interval = 2^(code+1) DWORDs)
//   pad_amt   : pad_amount code (amount = code+1 DWORDs), pad_enable=1
// ---------------------------------------------------------------------------
__device__ __forceinline__ void tdm_load_2d_bf16(const unsigned short* gsrc,
                                                 unsigned int lds_byte_addr,
                                                 unsigned int rows,
                                                 unsigned int rowlen,
                                                 unsigned int pad_iv,
                                                 unsigned int pad_amt) {
  unsigned long long ga = (unsigned long long)(size_t)gsrc;
  v4u g0;
  g0[0] = 1u;                                     // count=1, user mode
  g0[1] = lds_byte_addr;                          // lds_addr [63:32]
  g0[2] = (unsigned int)ga;                       // global_addr low
  g0[3] = ((unsigned int)(ga >> 32) & 0x01FFFFFFu) | (2u << 30);  // addr hi | type=2
  v8u g1;
  g1[0] = (1u << 16)                              // data_size = 2 bytes
        | (1u << 20)                              // pad_enable
        | (pad_iv << 22) | (pad_amt << 25);       // pad interval / amount
  g1[1] = (rowlen & 0xFFFFu) << 16;               // tensor_dim0 [79:48] low16
  g1[2] = (rowlen >> 16) | ((rows & 0xFFFFu) << 16);  // dim0 hi | tensor_dim1 lo
  g1[3] = (rows >> 16) | (rowlen << 16);          // dim1 hi | tile_dim0
  g1[4] = rows;                                   // tile_dim1 (tile_dim2 = 0)
  g1[5] = rowlen;                                 // tensor_dim0_stride low 32
  g1[6] = 0u;                                     // stride0 hi | stride1 lo
  g1[7] = 0u;
  // 2-D tensor: descriptor groups 2/3 are NULL (2-operand form)
  asm volatile("tensor_load_to_lds %0, %1" :: "s"(g0), "s"(g1) : "memory");
}

__device__ __forceinline__ unsigned int lds_offset_of(const void* p) {
  // low 32 bits of the flat shared-aperture address == LDS byte offset
  return (unsigned int)(size_t)p;
}

// ---------------------------------------------------------------------------
// Kernel 1: f32 -> bf16 convert (action_states)
// ---------------------------------------------------------------------------
__global__ void cvt_f32_bf16(const float* __restrict__ in,
                             unsigned short* __restrict__ out, int n) {
  int i = blockIdx.x * blockDim.x + threadIdx.x;
  if (i < n) out[i] = f32_to_bf16(in[i]);
}

// ---------------------------------------------------------------------------
// Kernel 2: W (D,DK) -> W^T (DK,D) in bf16 (both Wq and Wk via blockIdx.y)
// ---------------------------------------------------------------------------
__global__ void cvt_w_transpose(const float* __restrict__ Wq,
                                const float* __restrict__ Wk,
                                unsigned short* __restrict__ WqT,
                                unsigned short* __restrict__ WkT) {
  int i = blockIdx.x * blockDim.x + threadIdx.x;  // over DK*D
  if (i >= DKK * DD) return;
  int j = i / DD;         // dk index (output row)
  int d = i - j * DD;     // d index  (output col)
  const float* W = blockIdx.y ? Wk : Wq;
  unsigned short* WT = blockIdx.y ? WkT : WqT;
  WT[i] = f32_to_bf16(W[d * DKK + j]);
}

// ---------------------------------------------------------------------------
// Kernel 3: projection GEMM  Out(M,256) = A16(M,1024) * WT^T + bias
// A slab staged in LDS by the TDM; 8 waves x 2 col-tiles of WMMA
// LDS row stride 1056 ush (1024 + 2 pads of 16 ush), mid-row pad at k=512
// ---------------------------------------------------------------------------
#define ASLAB_STRIDE 1056

__global__ __launch_bounds__(256) void proj_kernel(
    const unsigned short* __restrict__ A16,   // (MTOT, 1024) bf16
    const unsigned short* __restrict__ WT,    // (256, 1024)  bf16 (K-major)
    const float* __restrict__ bias,           // (256)
    unsigned short* __restrict__ Out)         // (MTOT, 256)  bf16
{
  __shared__ unsigned short Aslab[16 * ASLAB_STRIDE];
  const int m0 = blockIdx.x * 16;
  const int tid = threadIdx.x;

  if (tid < 32) {   // wave 0 issues the TDM DMA (wave-scalar op)
    tdm_load_2d_bf16(&A16[(size_t)m0 * 1024], lds_offset_of(Aslab),
                     /*rows=*/16, /*rowlen=*/1024,
                     /*pad_iv=*/7 /*256 dw*/, /*pad_amt=*/7 /*8 dw*/);
    __builtin_amdgcn_s_wait_tensorcnt(0);
  }
  __syncthreads();

  const int wave = tid >> 5;
  const int lane = tid & 31;
  for (int t = 0; t < 2; ++t) {
    const int n0 = (wave * 2 + t) * 16;
    v8f acc = {};
#pragma unroll
    for (int k0 = 0; k0 < 1024; k0 += 32) {
      const int koff = k0 + (k0 >= 512 ? 16 : 0);   // mid-row TDM pad
      v16bf a = load_a_frag(&Aslab[koff], ASLAB_STRIDE);
      v16bf b = load_b_frag(&WT[(size_t)n0 * 1024 + k0], 1024);
      acc = __builtin_amdgcn_wmma_f32_16x16x32_bf16(
          false, a, false, b, (short)0, acc, false, false);
    }
    const int n  = n0 + (lane & 15);        // C/D layout: N = lane&15
    const int mh = (lane >> 4) * 8;         // M = j + 8*(lane>>4)
    const float bn = bias[n];
#pragma unroll
    for (int j = 0; j < 8; ++j) {
      Out[(size_t)(m0 + mh + j) * 256 + n] = f32_to_bf16(acc[j] + bn);
    }
  }
}

// ---------------------------------------------------------------------------
// Kernel 4 (fused): scores tile (16 x 2048) in LDS -> top-32 threshold ->
// masked softmax -> final output.  1024 blocks (8 batch x 128 query tiles).
// Q tile staged by TDM; LDS row stride 272 ush (256 + 16 pad per row).
// ---------------------------------------------------------------------------
#define SC_STRIDE 2064   // 2048 + 16 floats pad (bank stagger)
#define QT_STRIDE 272

__global__ __launch_bounds__(256) void scores_kernel(
    const unsigned short* __restrict__ Q16,  // (B*N, 256) bf16
    const unsigned short* __restrict__ K16,  // (B*N, 256) bf16
    float* __restrict__ out)                 // (B, N, N) f32
{
  extern __shared__ char smem_raw[];
  float*          sc   = (float*)smem_raw;                                  // 132096 B
  unsigned short* qt   = (unsigned short*)(smem_raw + 16 * SC_STRIDE * 4);  // 16*272*2 = 8704 B
  unsigned int*   hist = (unsigned int*)(smem_raw + 16 * SC_STRIDE * 4 + 16 * QT_STRIDE * 2);

  const int b   = blockIdx.x >> 7;
  const int m0  = (blockIdx.x & 127) * 16;
  const int tid = threadIdx.x;
  const int wave = tid >> 5;
  const int lane = tid & 31;

  const unsigned short* Qb = Q16 + (size_t)(b * NN + m0) * DKK;
  const unsigned short* Kb = K16 + (size_t)b * NN * DKK;

  if (tid < 32) {   // TDM stages the 16x256 Q tile
    tdm_load_2d_bf16(Qb, lds_offset_of(qt),
                     /*rows=*/16, /*rowlen=*/256,
                     /*pad_iv=*/6 /*128 dw*/, /*pad_amt=*/7 /*8 dw*/);
    __builtin_amdgcn_s_wait_tensorcnt(0);
  }
  __syncthreads();

  // scores: 128 key-tiles spread over 8 waves
  for (int it = 0; it < 16; ++it) {
    const int n0 = (wave + 8 * it) * 16;
    v8f acc = {};
#pragma unroll
    for (int k0 = 0; k0 < DKK; k0 += 32) {
      v16bf a = load_a_frag(&qt[k0], QT_STRIDE);
      v16bf kb = load_b_frag(&Kb[(size_t)n0 * 256 + k0], 256);
      acc = __builtin_amdgcn_wmma_f32_16x16x32_bf16(
          false, a, false, kb, (short)0, acc, false, false);
    }
    const int n  = n0 + (lane & 15);
    const int mh = (lane >> 4) * 8;
#pragma unroll
    for (int j = 0; j < 8; ++j)
      sc[(mh + j) * SC_STRIDE + n] = acc[j] * INV_SCALE;
  }
  __syncthreads();

  // top-32 + softmax: wave w handles rows 2w and 2w+1 (uniform control flow)
  for (int rr = 0; rr < 2; ++rr) {
    const int row = wave * 2 + rr;
    const float* r = &sc[row * SC_STRIDE];

    // row max (each lane covers 64 strided elements)
    float mx = -3.4e38f;
    for (int i = 0; i < 64; ++i) mx = fmaxf(mx, r[lane + 32 * i]);
    for (int o = 16; o > 0; o >>= 1) mx = fmaxf(mx, __shfl_xor(mx, o, 32));

    // radix-256 select: exact bit pattern of the 32nd-largest key
    unsigned int prefix = 0;
    unsigned int kk = TOPK;
    unsigned int* h = &hist[wave * 256];
    for (int pass = 0; pass < 4; ++pass) {
      const int shift = 24 - 8 * pass;
      const unsigned int pmask = (pass == 0) ? 0u : (0xFFFFFFFFu << (shift + 8));
#pragma unroll
      for (int j = 0; j < 8; ++j) h[lane * 8 + j] = 0;
      __syncthreads();
      for (int i = 0; i < 64; ++i) {
        unsigned int key = order_key(r[lane + 32 * i]);
        if ((key & pmask) == prefix)
          atomicAdd(&h[(key >> shift) & 255u], 1u);
      }
      __syncthreads();
      // per-lane bin sums + suffix scan across lanes (descending bins)
      unsigned int s = 0;
#pragma unroll
      for (int j = 0; j < 8; ++j) s += h[lane * 8 + j];
      unsigned int suf = s;
      for (int o = 1; o < 32; o <<= 1) {
        unsigned int t2 = __shfl_down(suf, o, 32);
        if (lane + o < 32) suf += t2;
      }
      unsigned long long ball = __ballot(suf >= kk);   // lanes <= L are true
      int lstar = 63 - __clzll(ball);
      unsigned int suf_star = __shfl(suf, lstar, 32);
      unsigned int s_star   = __shfl(s, lstar, 32);
      unsigned int cum = suf_star - s_star;            // count in higher lanes
      for (int j = 7; j >= 0; --j) {
        unsigned int c = h[lstar * 8 + j];
        if (cum + c >= kk) {
          prefix |= ((unsigned int)(lstar * 8 + j)) << shift;
          kk -= cum;
          break;
        }
        cum += c;
      }
      __syncthreads();
    }
    const unsigned int T = prefix;   // selected iff key >= T

    // softmax denominator over selected entries
    float sum = 0.0f;
    for (int i = 0; i < 64; ++i) {
      float v = r[lane + 32 * i];
      if (order_key(v) >= T) sum += expf(v - mx);
    }
    for (int o = 16; o > 0; o >>= 1) sum += __shfl_xor(sum, o, 32);
    const float inv = 1.0f / sum;

    float* orow = out + ((size_t)(b * NN + m0 + row)) * NN;
    for (int i = 0; i < 64; ++i) {
      const int col = lane + 32 * i;
      float v = r[col];
      orow[col] = (order_key(v) >= T) ? expf(v - mx) * inv : 0.0f;
    }
  }
}

// ---------------------------------------------------------------------------
// Launch
// ---------------------------------------------------------------------------
extern "C" void kernel_launch(void* const* d_in, const int* in_sizes, int n_in,
                              void* d_out, int out_size, void* d_ws, size_t ws_size,
                              hipStream_t stream) {
  (void)in_sizes; (void)n_in; (void)out_size; (void)ws_size;
  const float* A  = (const float*)d_in[0];
  const float* Wq = (const float*)d_in[1];
  const float* bq = (const float*)d_in[2];
  const float* Wk = (const float*)d_in[3];
  const float* bk = (const float*)d_in[4];
  float* out = (float*)d_out;

  // workspace layout (bf16 as ushort): ~49 MB total
  unsigned short* A16 = (unsigned short*)d_ws;
  unsigned short* WqT = A16 + (size_t)MTOT * DD;
  unsigned short* WkT = WqT + (size_t)DKK * DD;
  unsigned short* Q16 = WkT + (size_t)DKK * DD;
  unsigned short* K16 = Q16 + (size_t)MTOT * DKK;

  const int scores_lds = 16 * SC_STRIDE * 4 + 16 * QT_STRIDE * 2 + 8 * 256 * 4; // 148992 B
  hipFuncSetAttribute((const void*)scores_kernel,
                      hipFuncAttributeMaxDynamicSharedMemorySize, scores_lds);

  cvt_f32_bf16<<<(MTOT * DD) / 256, 256, 0, stream>>>(A, A16, MTOT * DD);

  dim3 gw((DKK * DD + 255) / 256, 2);
  cvt_w_transpose<<<gw, 256, 0, stream>>>(Wq, Wk, WqT, WkT);

  proj_kernel<<<MTOT / 16, 256, 0, stream>>>(A16, WqT, bq, Q16);
  proj_kernel<<<MTOT / 16, 256, 0, stream>>>(A16, WkT, bk, K16);

  scores_kernel<<<BB * (NN / 16), 256, scores_lds, stream>>>(Q16, K16, out);
}
